// GraphEncoder_65283502899367
// MI455X (gfx1250) — compile-verified, matrix operand order checked by reading, not run on hardware
//
#include <hip/hip_runtime.h>
#include <cstdint>
#include <cstddef>

// ---------------------------------------------------------------------------
// Types for CDNA5 WMMA (wave32): 16x16x32 bf16 -> f32 accumulate
// ---------------------------------------------------------------------------
typedef __attribute__((ext_vector_type(16))) __bf16 v16bf;
typedef __attribute__((ext_vector_type(8)))  float  v8f;

#define TPB 256

__device__ __forceinline__ unsigned short f2bfbits(float f) {
  unsigned u = __float_as_uint(f);
  u += 0x7fffu + ((u >> 16) & 1u);      // round-to-nearest-even
  return (unsigned short)(u >> 16);
}
__device__ __forceinline__ unsigned f32_order(float f) {
  unsigned u = __float_as_uint(f);
  return (u & 0x80000000u) ? ~u : (u | 0x80000000u);
}
__device__ __forceinline__ float f32_unorder(unsigned k) {
  unsigned u = (k & 0x80000000u) ? (k ^ 0x80000000u) : ~k;
  return __uint_as_float(u);
}

// ---------------------------------------------------------------------------
// Utility fills / conversions
// ---------------------------------------------------------------------------
__global__ void k_fill_f32(float* p, float v, long long n) {
  long long i = (long long)blockIdx.x * blockDim.x + threadIdx.x;
  if (i < n) p[i] = v;
}
__global__ void k_fill_u32(unsigned* p, unsigned v, long long n) {
  long long i = (long long)blockIdx.x * blockDim.x + threadIdx.x;
  if (i < n) p[i] = v;
}
__global__ void k_cvt_bf16(const float* __restrict__ x, unsigned short* __restrict__ o,
                           long long n) {
  long long i = (long long)blockIdx.x * blockDim.x + threadIdx.x;
  if (i < n) o[i] = f2bfbits(x[i]);
}

// ---------------------------------------------------------------------------
// LayerNorm (no affine, eps=1e-5), one wave32 per row, output bf16
// D must be a multiple of 32 (128 or 32 here)
// ---------------------------------------------------------------------------
__global__ void k_layernorm_bf16(const float* __restrict__ x, unsigned short* __restrict__ h,
                                 int n, int D) {
  int wid = blockIdx.x * (blockDim.x >> 5) + (threadIdx.x >> 5);
  if (wid >= n) return;
  int lane = threadIdx.x & 31;
  const float* row = x + (size_t)wid * D;
  int per = D >> 5;
  float s = 0.f;
  for (int i = 0; i < per; ++i) s += row[lane + 32 * i];
  for (int off = 16; off; off >>= 1) s += __shfl_xor(s, off, 32);
  float mu = s / (float)D;
  float var = 0.f;
  for (int i = 0; i < per; ++i) { float t = row[lane + 32 * i] - mu; var += t * t; }
  for (int off = 16; off; off >>= 1) var += __shfl_xor(var, off, 32);
  float inv = rsqrtf(var / (float)D + 1e-5f);
  unsigned short* orow = h + (size_t)wid * D;
  for (int i = 0; i < per; ++i)
    orow[lane + 32 * i] = f2bfbits((row[lane + 32 * i] - mu) * inv);
}

// ---------------------------------------------------------------------------
// Pack weight W[Do x Di] (row-major) into FRAGMENT-READY B layout:
// For column-tile jt (16 cols) and 32-wide k-block kb32, lane l holds its 16
// bf16 B-fragment values contiguously (32 bytes):
//   element e (vv=e>>1, hi=e&1):
//     k   = kb32*32 + ((vv&4)?16:0) + (l>>4)*8 + (vv&3)*2 + hi
//     col = jt*16 + (l&15);   value = W[col, k]
// Address: Bf[(((jtOff+jt)*nkb + kb32)*32 + l)*16 + e]
// ---------------------------------------------------------------------------
__global__ void k_pack_wfrag(const float* __restrict__ W, unsigned short* __restrict__ Bf,
                             int Di, int Do, int nkb, int jtOff) {
  int i = blockIdx.x * blockDim.x + threadIdx.x;
  if (i >= Do * Di) return;           // Do/16 * nkb * 32 * 16 == Do*Di
  int e = i & 15;
  int l = (i >> 4) & 31;
  int rest = i >> 9;
  int kb32 = rest % nkb;
  int jtl  = rest / nkb;
  int vv = e >> 1, hi = e & 1;
  int k = kb32 * 32 + ((vv & 4) ? 16 : 0) + ((l >> 4) * 8) + (vv & 3) * 2 + hi;
  int col = jtl * 16 + (l & 15);
  Bf[(((size_t)(jtOff + jtl) * nkb + kb32) * 32 + l) * 16 + e] =
      f2bfbits(W[(size_t)col * Di + k]);
}
__global__ void k_pack_bias(const float* __restrict__ b, float* __restrict__ biasOut,
                            int Do, int colOff) {
  int i = blockIdx.x * blockDim.x + threadIdx.x;
  if (i < Do) biasOut[colOff + i] = b[i];
}

// ---------------------------------------------------------------------------
// WMMA GEMM: out[M x Dcols] = A[M x K](bf16) * B(frag-packed bf16) + bias
// One 16x16 output tile per wave; 8 waves per block cover 8 column tiles.
// A fragment = two contiguous 16B runs per lane; B fragment = one 32B run.
// Epilogue: branch-free pointer-bump stores when this lane's 8 rows are all
// in-bounds (always true when nrows % 16 == 0); guarded fallback otherwise.
// ---------------------------------------------------------------------------
template<int K>
__global__ void k_gemm_bf16(const unsigned short* __restrict__ A,
                            const unsigned short* __restrict__ Bf,
                            const float* __restrict__ bias,
                            float* __restrict__ out,
                            int ldo, int Dcols, int nrows) {
  constexpr int NKB = K / 32;
  int wave = threadIdx.x >> 5;
  int lane = threadIdx.x & 31;
  int jt = blockIdx.y * 8 + wave;
  if (jt * 16 >= Dcols) return;               // wave-uniform: EXEC stays all-ones
  int r    = lane & 15;
  int half = lane >> 4;
  int row = blockIdx.x * 16 + r;
  if (row >= nrows) row = nrows - 1;          // address clamp only
  int col = jt * 16 + r;
  const unsigned short* arow  = A + (size_t)row * K;
  const unsigned short* bbase = Bf + ((size_t)jt * NKB * 32 + lane) * 16;

  v8f c = {};
#pragma unroll
  for (int kb32 = 0; kb32 < NKB; ++kb32) {
    union { v16bf vec; uint4 q[2]; } af, bfr;
    af.q[0]  = *(const uint4*)(arow + kb32 * 32 + half * 8);
    af.q[1]  = *(const uint4*)(arow + kb32 * 32 + 16 + half * 8);
    const uint4* bp = (const uint4*)(bbase + (size_t)kb32 * 32 * 16);
    bfr.q[0] = bp[0];
    bfr.q[1] = bp[1];
    c = __builtin_amdgcn_wmma_f32_16x16x32_bf16(false, af.vec, false, bfr.vec,
                                                (short)0, c, false, false);
  }
  float bj = bias[col];
  int baseRow = blockIdx.x * 16 + half * 8;   // D layout: VGPR vv -> M = half*8+vv
  float* po = out + (size_t)baseRow * ldo + col;
  if (baseRow + 8 <= nrows) {                 // common case: full 8-row strip
#pragma unroll
    for (int vv = 0; vv < 8; ++vv) {
      *po = c[vv] + bj;
      po += ldo;
    }
  } else {                                    // ragged tail (never hit if nrows%16==0)
#pragma unroll
    for (int vv = 0; vv < 8; ++vv) {
      if (baseRow + vv < nrows) po[(size_t)vv * ldo] = c[vv] + bj;
    }
  }
}

// ---------------------------------------------------------------------------
// Edge kernels (scatter softmax + message aggregation)
// ---------------------------------------------------------------------------
__global__ void k_edge_score(const float* __restrict__ proj, const int* __restrict__ src,
                             const int* __restrict__ dst, const float* __restrict__ eattr,
                             float* __restrict__ att, long long E, int ldo, int d,
                             int koff, int qoff, float scale) {
  long long e = (long long)blockIdx.x * blockDim.x + threadIdx.x;
  if (e >= E) return;
  const float4* kp = (const float4*)(proj + (size_t)src[e] * ldo + koff);
  const float4* qp = (const float4*)(proj + (size_t)dst[e] * ldo + qoff);
  float acc = 0.f;
  for (int i = 0; i < (d >> 2); ++i) {
    float4 a = kp[i], b = qp[i];
    acc += a.x * b.x + a.y * b.y + a.z * b.z + a.w * b.w;
  }
  att[e] = acc * scale * eattr[e];
}

__global__ void k_edge_segmax(const float* __restrict__ att, const int* __restrict__ dst,
                              unsigned* __restrict__ nmaxk, long long E) {
  long long e = (long long)blockIdx.x * blockDim.x + threadIdx.x;
  if (e >= E) return;
  atomicMax(&nmaxk[dst[e]], f32_order(att[e]));
}

__global__ void k_edge_expsum(float* __restrict__ att, const int* __restrict__ dst,
                              const unsigned* __restrict__ nmaxk, float* __restrict__ den,
                              long long E) {
  long long e = (long long)blockIdx.x * blockDim.x + threadIdx.x;
  if (e >= E) return;
  int t = dst[e];
  float ex = expf(att[e] - f32_unorder(nmaxk[t]));
  att[e] = ex;
  atomicAdd(&den[t], ex);
}

__global__ void k_edge_aggregate(const float* __restrict__ proj, const float* __restrict__ att,
                                 const float* __restrict__ den, const int* __restrict__ src,
                                 const int* __restrict__ dst, float* __restrict__ aggr,
                                 long long total, int ldo, int voff, int log2d) {
  long long i = (long long)blockIdx.x * blockDim.x + threadIdx.x;
  if (i >= total) return;
  long long e = i >> log2d;
  int c = (int)(i & ((1 << log2d) - 1));
  int t = dst[e], s = src[e];
  float a = att[e] / (den[t] + 1e-16f);
  atomicAdd(&aggr[((size_t)t << log2d) + c], a * proj[(size_t)s * ldo + voff + c]);
}

__global__ void k_node_finalize(const float* __restrict__ aggr, const float* __restrict__ proj,
                                float* __restrict__ xout, long long total, int ldo, int roff,
                                int log2d) {
  long long i = (long long)blockIdx.x * blockDim.x + threadIdx.x;
  if (i >= total) return;
  long long node = i >> log2d;
  int c = (int)(i & ((1 << log2d) - 1));
  float x = aggr[i];
  float g = 0.5f * x * (1.f + erff(x * 0.70710678118654752f));  // exact GELU
  xout[i] = g - proj[(size_t)node * ldo + roff + c];
}

// ---------------------------------------------------------------------------
// Global mean pool
// ---------------------------------------------------------------------------
__global__ void k_pool_accum(const float* __restrict__ x, const int* __restrict__ batch,
                             float* __restrict__ sums, float* __restrict__ cnt,
                             long long total, int log2d) {
  long long i = (long long)blockIdx.x * blockDim.x + threadIdx.x;
  if (i >= total) return;
  long long node = i >> log2d;
  int c = (int)(i & ((1 << log2d) - 1));
  int g = batch[node];
  atomicAdd(&sums[((size_t)g << log2d) + c], x[i]);
  if (c == 0) atomicAdd(&cnt[g], 1.f);
}

__global__ void k_pool_div(const float* __restrict__ sums, const float* __restrict__ cnt,
                           float* __restrict__ out, long long total, int log2d) {
  long long i = (long long)blockIdx.x * blockDim.x + threadIdx.x;
  if (i >= total) return;
  out[i] = sums[i] / fmaxf(cnt[i >> log2d], 1.f);
}

// ---------------------------------------------------------------------------
// Host orchestration
// ---------------------------------------------------------------------------
extern "C" void kernel_launch(void* const* d_in, const int* in_sizes, int n_in,
                              void* d_out, int out_size, void* d_ws, size_t ws_size,
                              hipStream_t stream) {
  (void)n_in; (void)ws_size;
  const float* x     = (const float*)d_in[0];
  const int*   eidx  = (const int*)d_in[1];
  const float* eattr = (const float*)d_in[2];
  const int*   batch = (const int*)d_in[3];
  const float* wk1 = (const float*)d_in[4];  const float* bk1 = (const float*)d_in[5];
  const float* wq1 = (const float*)d_in[6];  const float* bq1 = (const float*)d_in[7];
  const float* wv1 = (const float*)d_in[8];  const float* bv1 = (const float*)d_in[9];
  const float* wr1 = (const float*)d_in[10]; const float* br1 = (const float*)d_in[11];
  const float* wk2 = (const float*)d_in[12]; const float* bk2 = (const float*)d_in[13];
  const float* wq2 = (const float*)d_in[14]; const float* bq2 = (const float*)d_in[15];
  const float* wv2 = (const float*)d_in[16]; const float* bv2 = (const float*)d_in[17];
  const float* wr2 = (const float*)d_in[18]; const float* br2 = (const float*)d_in[19];
  float* out = (float*)d_out;

  const int FIN = 128;
  const long long N = in_sizes[0] / FIN;      // 100000
  const long long E = in_sizes[1] / 2;        // 1600000
  const int H1   = in_sizes[4] / FIN;         // 32
  const int DOUT = in_sizes[12] / H1;         // 64
  const int G    = out_size / DOUT;           // 128
  const int* src = eidx;
  const int* dst = eidx + E;

  auto ilog2 = [](int v) { int l = 0; while ((1 << l) < v) ++l; return l; };
  const int l2H1 = ilog2(H1), l2DO = ilog2(DOUT);

  // ---- workspace carve-out (all from d_ws) ----
  char* ws = (char*)d_ws;
  size_t off = 0;
  auto take = [&](size_t bytes) -> void* {
    void* p = ws + off;
    off += (bytes + 255) & ~(size_t)255;
    return p;
  };
  unsigned short* hbf  = (unsigned short*)take((size_t)N * FIN * 2); // LN output, bf16
  unsigned short* xbf  = (unsigned short*)take((size_t)N * FIN * 2); // raw input, bf16 (residual)
  unsigned short* wbfA = (unsigned short*)take((size_t)FIN * 3 * DOUT * 2); // frag-packed KQV
  unsigned short* wbfR = (unsigned short*)take((size_t)FIN * DOUT * 2);     // frag-packed R
  float* biasA = (float*)take((size_t)3 * DOUT * 4);
  float* biasR = (float*)take((size_t)DOUT * 4);
  float* proj  = (float*)take((size_t)N * 4 * DOUT * 4);  // N x 256 max (K|Q|V|R)
  float* att   = (float*)take((size_t)E * 4);
  unsigned* nmaxk = (unsigned*)take((size_t)N * 4);
  float* den   = (float*)take((size_t)N * 4);
  float* aggr  = (float*)take((size_t)N * DOUT * 4);
  float* x1    = (float*)take((size_t)N * H1 * 4);
  float* x2    = (float*)take((size_t)N * DOUT * 4);
  float* sums  = (float*)take((size_t)G * DOUT * 4);
  float* cnt   = (float*)take((size_t)G * 4);

  auto b1 = [](long long n) { return dim3((unsigned)((n + TPB - 1) / TPB)); };
  const unsigned MT = (unsigned)((N + 15) / 16);          // M tiles for WMMA GEMM
  const dim3 LNG((unsigned)((N + 7) / 8));                // 8 rows (waves) per block

  // =================== Layer 1 (FIN=128 -> H1=32) ===================
  {
    const int d = H1, Dc = 3 * d, ldo = 4 * d;     // 96, 128
    const int nkb = FIN / 32;                      // 4
    k_layernorm_bf16<<<LNG, TPB, 0, stream>>>(x, hbf, (int)N, FIN);
    k_cvt_bf16<<<b1(N * FIN), TPB, 0, stream>>>(x, xbf, N * FIN);
    k_pack_wfrag<<<b1((long long)FIN * d), TPB, 0, stream>>>(wk1, wbfA, FIN, d, nkb, 0);
    k_pack_wfrag<<<b1((long long)FIN * d), TPB, 0, stream>>>(wq1, wbfA, FIN, d, nkb, d / 16);
    k_pack_wfrag<<<b1((long long)FIN * d), TPB, 0, stream>>>(wv1, wbfA, FIN, d, nkb, 2 * d / 16);
    k_pack_wfrag<<<b1((long long)FIN * d), TPB, 0, stream>>>(wr1, wbfR, FIN, d, nkb, 0);
    k_pack_bias<<<b1(d), TPB, 0, stream>>>(bk1, biasA, d, 0);
    k_pack_bias<<<b1(d), TPB, 0, stream>>>(bq1, biasA, d, d);
    k_pack_bias<<<b1(d), TPB, 0, stream>>>(bv1, biasA, d, 2 * d);
    k_pack_bias<<<b1(d), TPB, 0, stream>>>(br1, biasR, d, 0);
    // KQV from LayerNormed h; residual R from raw x
    k_gemm_bf16<128><<<dim3(MT, (unsigned)((Dc / 16 + 7) / 8)), TPB, 0, stream>>>(
        hbf, wbfA, biasA, proj, ldo, Dc, (int)N);
    k_gemm_bf16<128><<<dim3(MT, 1), TPB, 0, stream>>>(
        xbf, wbfR, biasR, proj + 3 * d, ldo, d, (int)N);

    k_fill_u32<<<b1(N), TPB, 0, stream>>>(nmaxk, 0u, N);
    k_fill_f32<<<b1(N), TPB, 0, stream>>>(den, 0.f, N);
    k_fill_f32<<<b1(N * d), TPB, 0, stream>>>(aggr, 0.f, N * d);
    k_edge_score<<<b1(E), TPB, 0, stream>>>(proj, src, dst, eattr, att, E, ldo, d, 0, d,
                                            1.f / sqrtf((float)d));
    k_edge_segmax<<<b1(E), TPB, 0, stream>>>(att, dst, nmaxk, E);
    k_edge_expsum<<<b1(E), TPB, 0, stream>>>(att, dst, nmaxk, den, E);
    k_edge_aggregate<<<b1(E * d), TPB, 0, stream>>>(proj, att, den, src, dst, aggr,
                                                    E * d, ldo, 2 * d, l2H1);
    k_node_finalize<<<b1(N * d), TPB, 0, stream>>>(aggr, proj, x1, N * d, ldo, 3 * d, l2H1);
  }

  // =================== Layer 2 (H1=32 -> DOUT=64) ===================
  {
    const int d = DOUT, Din = H1, Dc = 3 * d, ldo = 4 * d;  // 192, 256
    const int nkb = Din / 32;                               // 1
    k_layernorm_bf16<<<LNG, TPB, 0, stream>>>(x1, hbf, (int)N, Din);
    k_cvt_bf16<<<b1(N * Din), TPB, 0, stream>>>(x1, xbf, N * Din);
    k_pack_wfrag<<<b1((long long)Din * d), TPB, 0, stream>>>(wk2, wbfA, Din, d, nkb, 0);
    k_pack_wfrag<<<b1((long long)Din * d), TPB, 0, stream>>>(wq2, wbfA, Din, d, nkb, d / 16);
    k_pack_wfrag<<<b1((long long)Din * d), TPB, 0, stream>>>(wv2, wbfA, Din, d, nkb, 2 * d / 16);
    k_pack_wfrag<<<b1((long long)Din * d), TPB, 0, stream>>>(wr2, wbfR, Din, d, nkb, 0);
    k_pack_bias<<<b1(d), TPB, 0, stream>>>(bk2, biasA, d, 0);
    k_pack_bias<<<b1(d), TPB, 0, stream>>>(bq2, biasA, d, d);
    k_pack_bias<<<b1(d), TPB, 0, stream>>>(bv2, biasA, d, 2 * d);
    k_pack_bias<<<b1(d), TPB, 0, stream>>>(br2, biasR, d, 0);
    k_gemm_bf16<32><<<dim3(MT, (unsigned)((Dc / 16 + 7) / 8)), TPB, 0, stream>>>(
        hbf, wbfA, biasA, proj, ldo, Dc, (int)N);
    k_gemm_bf16<32><<<dim3(MT, 1), TPB, 0, stream>>>(
        xbf, wbfR, biasR, proj + 3 * d, ldo, d, (int)N);

    k_fill_u32<<<b1(N), TPB, 0, stream>>>(nmaxk, 0u, N);
    k_fill_f32<<<b1(N), TPB, 0, stream>>>(den, 0.f, N);
    k_fill_f32<<<b1(N * d), TPB, 0, stream>>>(aggr, 0.f, N * d);
    k_edge_score<<<b1(E), TPB, 0, stream>>>(proj, src, dst, eattr, att, E, ldo, d, 0, d,
                                            1.f / sqrtf((float)d));
    k_edge_segmax<<<b1(E), TPB, 0, stream>>>(att, dst, nmaxk, E);
    k_edge_expsum<<<b1(E), TPB, 0, stream>>>(att, dst, nmaxk, den, E);
    k_edge_aggregate<<<b1(E * d), TPB, 0, stream>>>(proj, att, den, src, dst, aggr,
                                                    E * d, ldo, 2 * d, l2DO);
    k_node_finalize<<<b1(N * d), TPB, 0, stream>>>(aggr, proj, x2, N * d, ldo, 3 * d, l2DO);
  }

  // =================== Global mean pool ===================
  k_fill_f32<<<b1((long long)G * DOUT), TPB, 0, stream>>>(sums, 0.f, (long long)G * DOUT);
  k_fill_f32<<<b1(G), TPB, 0, stream>>>(cnt, 0.f, G);
  k_pool_accum<<<b1(N * DOUT), TPB, 0, stream>>>(x2, batch, sums, cnt, N * DOUT, l2DO);
  k_pool_div<<<b1((long long)G * DOUT), TPB, 0, stream>>>(sums, cnt, out, (long long)G * DOUT, l2DO);
}
